// RumorDetect_73607149519596
// MI455X (gfx1250) — matrix operand order repaired and checked by reading, not compiled, not run on hardware
//
#include <hip/hip_runtime.h>
#include <hip/hip_bf16.h>

// ---------------------------------------------------------------------------
// Problem constants (from reference)
// ---------------------------------------------------------------------------
constexpr int gB = 4;
constexpr int gN = 4096;
constexpr int gNFEAT = 128;
constexpr int gNHID = 64;
constexpr int gNCLASS = 64;
constexpr int gHIDDEN = 32;
constexpr int gNC = 25;
constexpr int gDMID = 4096;
constexpr float gEPS = 1e-5f;

// Output layout (flat, return order): C_s(4*25) | T_mask(4*4096) | c_id(4*2) | out(4*4096*2) | adj(4*4096*4096)
constexpr size_t OFF_CS   = 0;
constexpr size_t OFF_TM   = 100;
constexpr size_t OFF_CID  = 16484;
constexpr size_t OFF_OUT  = 16492;
constexpr size_t OFF_ADJ  = 49260;

typedef __bf16 bf16;
typedef __attribute__((ext_vector_type(8)))  __bf16 bf16x8;
typedef __attribute__((ext_vector_type(16))) __bf16 v16bf;
typedef __attribute__((ext_vector_type(8)))  float  v8f;
typedef __attribute__((ext_vector_type(4)))  int    v4i;

#define AS1 __attribute__((address_space(1)))
#define AS3 __attribute__((address_space(3)))

#if __has_builtin(__builtin_amdgcn_global_load_async_to_lds_b128)
#define HAS_ASYNC_LDS 1
#else
#define HAS_ASYNC_LDS 0
#endif

__device__ inline void async_copy_b128(const void* g, void* l) {
#if HAS_ASYNC_LDS
    __builtin_amdgcn_global_load_async_to_lds_b128((AS1 v4i*)g, (AS3 v4i*)l, 0, 0);
#else
    *(bf16x8*)l = *(const bf16x8*)g;
#endif
}
__device__ inline void wait_async0() {
#if HAS_ASYNC_LDS
#if __has_builtin(__builtin_amdgcn_s_wait_asynccnt)
    __builtin_amdgcn_s_wait_asynccnt(0);
#else
    asm volatile("s_wait_asynccnt 0x0" ::: "memory");
#endif
#endif
}

// ---------------------------------------------------------------------------
// WMMA fragment loaders (CDNA5 wave32 layouts, 16x16x32 bf16)
//  A (16x32): lane l -> row (l&15); half=(l>>4). VGPR v0..3 hold K=8h+0..7,
//             v4..7 hold K=16+8h+0..7  -> two contiguous 16B reads.
//  B (32x16): lane l -> col (l&15); same K order as A (8h+0..7, 16+8h+0..7).
// ---------------------------------------------------------------------------
__device__ inline v16bf load_frag_a(const bf16* row32, int hlf) {
    bf16x8 lo = *(const bf16x8*)(row32 + hlf * 8);
    bf16x8 hi = *(const bf16x8*)(row32 + 16 + hlf * 8);
    v16bf f;
#pragma unroll
    for (int i = 0; i < 8; ++i) { f[i] = lo[i]; f[i + 8] = hi[i]; }
    return f;
}
__device__ inline v16bf load_frag_b(const bf16* col32, int hlf) {
    bf16x8 lo = *(const bf16x8*)(col32 + hlf * 8);
    bf16x8 hi = *(const bf16x8*)(col32 + 16 + hlf * 8);
    v16bf f;
#pragma unroll
    for (int i = 0; i < 8; ++i) { f[i] = lo[i]; f[i + 8] = hi[i]; }
    return f;
}

// ---------------------------------------------------------------------------
// adj = relu(a1+a2+a3)  (one float4 per thread)
// ---------------------------------------------------------------------------
__global__ void adj_relu_kernel(const float* __restrict__ a1, const float* __restrict__ a2,
                                const float* __restrict__ a3, float* __restrict__ adj, long total4) {
    long i = (long)blockIdx.x * blockDim.x + threadIdx.x;
    if (i >= total4) return;
    float4 v1 = ((const float4*)a1)[i];
    float4 v2 = ((const float4*)a2)[i];
    float4 v3 = ((const float4*)a3)[i];
    float4 o;
    o.x = fmaxf(v1.x + v2.x + v3.x, 0.f);
    o.y = fmaxf(v1.y + v2.y + v3.y, 0.f);
    o.z = fmaxf(v1.z + v2.z + v3.z, 0.f);
    o.w = fmaxf(v1.w + v2.w + v3.w, 0.f);
    ((float4*)adj)[i] = o;
}

// ---------------------------------------------------------------------------
// WMMA GEMM: Out[b] = act(adj[b] (NxN fp32) @ W[b] + bias)
// W supplied TRANSPOSED as BwT (TN x N bf16 per batch) so the LDS B-tile is
// K-contiguous and staged with GLOBAL_LOAD_ASYNC_TO_LDS_B128 (ASYNCcnt path).
// Block = 256 threads (8 waves). Wave grid (TM/16) x (TN/32); 2 WMMA/wave/K-step.
// ---------------------------------------------------------------------------
template <int TM, int TN>
__global__ __launch_bounds__(256) void adj_gemm_kernel(const float* __restrict__ Aadj,
                                                       const bf16* __restrict__ BwT,
                                                       const float* __restrict__ bias,
                                                       float* __restrict__ Outf, int relu) {
    constexpr int WM = TM / 16, WN = TN / 32;
    static_assert(WM * WN == 8, "8 waves");
    __shared__ __align__(16) bf16 As[TM * 32];
    __shared__ __align__(16) bf16 Bs[TN * 32];

    const int b = blockIdx.y;
    const int rowbase = blockIdx.x * TM;
    const float* A = Aadj + (size_t)b * gN * gN;
    const bf16* Bp = BwT + (size_t)b * TN * gN;   // [col][k] row-major, stride gN

    const int tid = threadIdx.x;
    const int lane = tid & 31, wave = tid >> 5;
    const int hlf = lane >> 4, l15 = lane & 15;
    const int wm = wave % WM, wn = wave / WM;

    v8f acc0 = {}; v8f acc1 = {};

    for (int kb = 0; kb < gN; kb += 32) {
        // stage B tile (bf16, K-contiguous) via async global->LDS copy
        if (tid < TN * 4) {
            int col = tid >> 2, ko = (tid & 3) * 8;
            async_copy_b128(&Bp[(size_t)col * gN + kb + ko], &Bs[col * 32 + ko]);
        }
        // stage A tile fp32 -> bf16 (VALU convert, ds_store)
        for (int e = tid * 4; e < TM * 32; e += 256 * 4) {
            int r = e >> 5, c = e & 31;
            float4 v = *(const float4*)(A + (size_t)(rowbase + r) * gN + kb + c);
            bf16* d = &As[r * 32 + c];
            d[0] = (bf16)v.x; d[1] = (bf16)v.y; d[2] = (bf16)v.z; d[3] = (bf16)v.w;
        }
        // prefetch next A tile rows into cache (global_prefetch_b8)
        if (kb + 32 < gN && tid < TM)
            __builtin_prefetch(A + (size_t)(rowbase + tid) * gN + kb + 32, 0, 1);
        wait_async0();
        __syncthreads();

        v16bf af = load_frag_a(&As[(wm * 16 + l15) * 32], hlf);
        v16bf bf0 = load_frag_b(&Bs[(wn * 32 + l15) * 32], hlf);
        acc0 = __builtin_amdgcn_wmma_f32_16x16x32_bf16(false, af, false, bf0, (short)0, acc0, false, false);
        v16bf bf1 = load_frag_b(&Bs[(wn * 32 + 16 + l15) * 32], hlf);
        acc1 = __builtin_amdgcn_wmma_f32_16x16x32_bf16(false, af, false, bf1, (short)0, acc1, false, false);
        __syncthreads();
    }

    const int c0 = wn * 32 + l15, c1 = c0 + 16;
    const float bi0 = bias[c0], bi1 = bias[c1];
#pragma unroll
    for (int v = 0; v < 8; ++v) {
        int rg = rowbase + wm * 16 + v + 8 * hlf;
        float x0 = acc0[v] + bi0, x1 = acc1[v] + bi1;
        if (relu) { x0 = fmaxf(x0, 0.f); x1 = fmaxf(x1, 0.f); }
        Outf[((size_t)b * gN + rg) * TN + c0] = x0;
        Outf[((size_t)b * gN + rg) * TN + c1] = x1;
    }
}

// ---------------------------------------------------------------------------
// Fused: out[b,n] = sum_m relu(<h[b,n,:], h[b,m,:]>) * state[b,m]   (WMMA, S never stored)
// Block handles 128 rows; loops m in 64-chunks; K=64 -> 2 WMMA k-steps.
// H tiles staged with async global->LDS copies.
// ---------------------------------------------------------------------------
__global__ __launch_bounds__(256) void hht_state_kernel(const bf16* __restrict__ hb,
                                                        const float* __restrict__ state,
                                                        float* __restrict__ outvec) {
    __shared__ __align__(16) bf16 Hn[128 * 64];
    __shared__ __align__(16) bf16 Hm[64 * 64];
    __shared__ float Sst[64];

    const int b = blockIdx.y;
    const int nbase = blockIdx.x * 128;
    const int tid = threadIdx.x;
    const int lane = tid & 31, wave = tid >> 5;
    const int hlf = lane >> 4, l15 = lane & 15;
    const bf16* hB = hb + (size_t)b * gN * 64;

    for (int e = tid * 8; e < 128 * 64; e += 256 * 8)
        async_copy_b128(&hB[(size_t)nbase * 64 + e], &Hn[e]);

    float rsum[8] = {};
    for (int mb = 0; mb < gN; mb += 64) {
        wait_async0();
        __syncthreads();
        for (int e = tid * 8; e < 64 * 64; e += 256 * 8)
            async_copy_b128(&hB[(size_t)mb * 64 + e], &Hm[e]);
        if (tid < 64) Sst[tid] = state[(size_t)b * gN + mb + tid];
        wait_async0();
        __syncthreads();

        v8f acc[4] = {v8f{}, v8f{}, v8f{}, v8f{}};
#pragma unroll
        for (int ks = 0; ks < 2; ++ks) {
            v16bf af = load_frag_a(&Hn[(wave * 16 + l15) * 64 + ks * 32], hlf);
#pragma unroll
            for (int t = 0; t < 4; ++t) {
                v16bf bf = load_frag_b(&Hm[(t * 16 + l15) * 64 + ks * 32], hlf);
                acc[t] = __builtin_amdgcn_wmma_f32_16x16x32_bf16(false, af, false, bf, (short)0, acc[t], false, false);
            }
        }
#pragma unroll
        for (int t = 0; t < 4; ++t) {
            float sv = Sst[t * 16 + l15];
#pragma unroll
            for (int v = 0; v < 8; ++v) rsum[v] += fmaxf(acc[t][v], 0.f) * sv;
        }
    }
    // reduce each output row (spread over 16 lanes)
#pragma unroll
    for (int v = 0; v < 8; ++v) {
        float s = rsum[v];
        for (int m = 8; m >= 1; m >>= 1) s += __shfl_xor(s, m, 16);
        rsum[v] = s;
    }
    if (l15 == 0) {
#pragma unroll
        for (int v = 0; v < 8; ++v)
            outvec[(size_t)b * gN + nbase + wave * 16 + v + 8 * hlf] = rsum[v];
    }
}

// ---------------------------------------------------------------------------
// Small per-node GEMM: act(in[b,n,:K] @ w[K,C] (+ bias))
// Optional outputs: row-major f32 (B,n,C) and TRANSPOSED bf16 (B,C,n) for the
// WMMA GEMM B-operand.
// ---------------------------------------------------------------------------
__global__ void small_gemm_kernel(const float* __restrict__ in, const float* __restrict__ w,
                                  const float* __restrict__ bias, float* __restrict__ outf,
                                  bf16* __restrict__ outbT, int n, int k, int c, int relu) {
    long t = (long)blockIdx.x * blockDim.x + threadIdx.x;
    long total = (long)gB * n * c;
    if (t >= total) return;
    int col = (int)(t % c);
    long nn = t / c;
    int bidx = (int)(nn / n);
    int row = (int)(nn % n);
    const float* xr = in + nn * k;
    float acc = bias ? bias[col] : 0.f;
    for (int kk = 0; kk < k; ++kk) acc += xr[kk] * w[(long)kk * c + col];
    if (relu) acc = fmaxf(acc, 0.f);
    if (outf) outf[t] = acc;
    if (outbT) outbT[((size_t)bidx * c + col) * n + row] = (bf16)acc;
}

// log_softmax over last dim (c) of (B, n, c), in place
__global__ void log_softmax_kernel(float* __restrict__ h, int n, int c) {
    long t = (long)blockIdx.x * blockDim.x + threadIdx.x;
    if (t >= (long)gB * n) return;
    float* r = h + t * c;
    float m = r[0];
    for (int i = 1; i < c; ++i) m = fmaxf(m, r[i]);
    float s = 0.f;
    for (int i = 0; i < c; ++i) s += expf(r[i] - m);
    float ls = m + logf(s);
    for (int i = 0; i < c; ++i) r[i] = r[i] - ls;
}

// BatchNorm over node axis: stats per (batch, col) over rows
__global__ void bn_stats_kernel(const float* __restrict__ x, float* __restrict__ mu,
                                float* __restrict__ istd, int rows, int cols) {
    int c = blockIdx.x, b = blockIdx.y;
    const float* base = x + (size_t)b * rows * cols;
    float s = 0.f, ss = 0.f;
    for (int r = threadIdx.x; r < rows; r += blockDim.x) {
        float v = base[(size_t)r * cols + c];
        s += v; ss += v * v;
    }
    __shared__ float S1[256], S2[256];
    S1[threadIdx.x] = s; S2[threadIdx.x] = ss;
    __syncthreads();
    for (int o = 128; o > 0; o >>= 1) {
        if (threadIdx.x < o) { S1[threadIdx.x] += S1[threadIdx.x + o]; S2[threadIdx.x] += S2[threadIdx.x + o]; }
        __syncthreads();
    }
    if (threadIdx.x == 0) {
        float m = S1[0] / rows;
        float var = S2[0] / rows - m * m;
        mu[b * cols + c] = m;
        istd[b * cols + c] = rsqrtf(var + gEPS);
    }
}

__global__ void bn_apply_kernel(float* __restrict__ x, bf16* __restrict__ xb,
                                const float* __restrict__ mu, const float* __restrict__ istd,
                                const float* __restrict__ g, const float* __restrict__ bb,
                                int rows, int cols) {
    long t = (long)blockIdx.x * blockDim.x + threadIdx.x;
    long total = (long)gB * rows * cols;
    if (t >= total) return;
    int c = (int)(t % cols);
    int b = (int)(t / ((long)rows * cols));
    float v = (x[t] - mu[b * cols + c]) * istd[b * cols + c] * g[c] + bb[c];
    x[t] = v;
    if (xb) xb[t] = (bf16)v;
}

// argmax(25), centrality sigmoid, segment sums (atomics)
__global__ void cluster_kernel(const float* __restrict__ c2, const float* __restrict__ cen_w,
                               const float* __restrict__ cen_b, int* __restrict__ ids,
                               float* __restrict__ sums, float* __restrict__ cnts) {
    long t = (long)blockIdx.x * blockDim.x + threadIdx.x;
    if (t >= (long)gB * gN) return;
    int b = (int)(t / gN);
    const float* r = c2 + t * gNC;
    int best = 0; float bv = r[0]; float dot = r[0] * cen_w[0];
    for (int j = 1; j < gNC; ++j) {
        float v = r[j];
        if (v > bv) { bv = v; best = j; }
        dot += v * cen_w[j];
    }
    float ss = 1.f / (1.f + expf(-(dot + cen_b[0])));
    ids[t] = best;
    atomicAdd(&sums[b * gNC + best], ss);
    atomicAdd(&cnts[b * gNC + best], 1.f);
}

__global__ void score_top_kernel(const float* __restrict__ sums, const float* __restrict__ cnts,
                                 float* __restrict__ Cs, float* __restrict__ cid,
                                 int* __restrict__ topbuf) {
    int b = blockIdx.x;
    __shared__ float sc[gNC];
    if (threadIdx.x < gNC) {
        float cn = cnts[b * gNC + threadIdx.x];
        float v = sums[b * gNC + threadIdx.x] / fmaxf(cn, 1.f);
        sc[threadIdx.x] = v;
        Cs[b * gNC + threadIdx.x] = v;
    }
    __syncthreads();
    if (threadIdx.x == 0) {
        int t0 = 0; float v0 = sc[0];
        for (int j = 1; j < gNC; ++j) if (sc[j] > v0) { v0 = sc[j]; t0 = j; }
        int t1 = (t0 == 0) ? 1 : 0; float v1 = sc[t1];
        for (int j = 0; j < gNC; ++j) if (j != t0 && sc[j] > v1) { v1 = sc[j]; t1 = j; }
        topbuf[b * 2] = t0; topbuf[b * 2 + 1] = t1;
        cid[b * 2] = (float)t0; cid[b * 2 + 1] = (float)t1;
    }
}

__global__ void mask_kernel(const int* __restrict__ ids, const int* __restrict__ topbuf,
                            float* __restrict__ mask) {
    long t = (long)blockIdx.x * blockDim.x + threadIdx.x;
    if (t >= (long)gB * gN) return;
    int b = (int)(t / gN);
    int id = ids[t];
    mask[t] = (float)((id == topbuf[b * 2]) + (id == topbuf[b * 2 + 1]));
}

// BatchNorm over batch axis (B=4) for (B, cols), in place
__global__ void bn_batch_kernel(float* __restrict__ x, const float* __restrict__ g,
                                const float* __restrict__ bb, int cols) {
    int j = blockIdx.x * blockDim.x + threadIdx.x;
    if (j >= cols) return;
    float v0 = x[j], v1 = x[cols + j], v2 = x[2 * (size_t)cols + j], v3 = x[3 * (size_t)cols + j];
    float m = (v0 + v1 + v2 + v3) * 0.25f;
    float d0 = v0 - m, d1 = v1 - m, d2 = v2 - m, d3 = v3 - m;
    float var = (d0 * d0 + d1 * d1 + d2 * d2 + d3 * d3) * 0.25f;
    float is = rsqrtf(var + gEPS) * g[j];
    x[j] = d0 * is + bb[j];
    x[cols + j] = d1 * is + bb[j];
    x[2 * (size_t)cols + j] = d2 * is + bb[j];
    x[3 * (size_t)cols + j] = d3 * is + bb[j];
}

// (B,k) @ (k,c) + bias, act: 0 none, 1 relu, 2 sigmoid
__global__ void mlp_kernel(const float* __restrict__ in, const float* __restrict__ w,
                           const float* __restrict__ bias, float* __restrict__ out,
                           int k, int c, int act) {
    long t = (long)blockIdx.x * blockDim.x + threadIdx.x;
    if (t >= (long)gB * c) return;
    int j = (int)(t % c);
    int b = (int)(t / c);
    const float* xr = in + (long)b * k;
    float acc = bias[j];
    for (int kk = 0; kk < k; ++kk) acc += xr[kk] * w[(long)kk * c + j];
    if (act == 1) acc = fmaxf(acc, 0.f);
    else if (act == 2) acc = 1.f / (1.f + expf(-acc));
    out[t] = acc;
}

__global__ void zero_kernel(float* __restrict__ a, float* __restrict__ b, int n) {
    int i = blockIdx.x * blockDim.x + threadIdx.x;
    if (i < n) { a[i] = 0.f; b[i] = 0.f; }
}

// ---------------------------------------------------------------------------
extern "C" void kernel_launch(void* const* d_in, const int* in_sizes, int n_in,
                              void* d_out, int out_size, void* d_ws, size_t ws_size,
                              hipStream_t stream) {
    const float* x       = (const float*)d_in[0];
    const float* a1      = (const float*)d_in[1];
    const float* a2      = (const float*)d_in[2];
    const float* a3      = (const float*)d_in[3];
    const float* state   = (const float*)d_in[4];
    const float* gc1_w   = (const float*)d_in[5];
    const float* gc1_b   = (const float*)d_in[6];
    const float* gc2_w   = (const float*)d_in[7];
    const float* gc2_b   = (const float*)d_in[8];
    const float* norm_g  = (const float*)d_in[9];
    const float* norm_b  = (const float*)d_in[10];
    const float* clgc_w  = (const float*)d_in[11];
    const float* clgc_b  = (const float*)d_in[12];
    const float* clmlp_w = (const float*)d_in[13];
    const float* clmlp_b = (const float*)d_in[14];
    const float* cln_g   = (const float*)d_in[15];
    const float* cln_b   = (const float*)d_in[16];
    const float* cen_w   = (const float*)d_in[17];
    const float* cen_b   = (const float*)d_in[18];
    const float* norm1_g = (const float*)d_in[19];
    const float* norm1_b = (const float*)d_in[20];
    const float* mlp1_w  = (const float*)d_in[21];
    const float* mlp1_b  = (const float*)d_in[22];
    const float* norm2_g = (const float*)d_in[23];
    const float* norm2_b = (const float*)d_in[24];
    const float* mlp2_w  = (const float*)d_in[25];
    const float* mlp2_b  = (const float*)d_in[26];

    float* out   = (float*)d_out;
    float* Cs    = out + OFF_CS;
    float* Tm    = out + OFF_TM;
    float* cid   = out + OFF_CID;
    float* outp  = out + OFF_OUT;
    float* adj   = out + OFF_ADJ;

    // workspace bump allocator
    char* base = (char*)d_ws;
    size_t off = 0;
    auto alloc = [&](size_t bytes) -> void* {
        off = (off + 255) & ~(size_t)255;
        void* p = base + off;
        off += bytes;
        return p;
    };
    bf16*  xw1bT   = (bf16*)alloc((size_t)gB * gN * 64 * 2);   // (B, 64, N)
    bf16*  hw2bT   = (bf16*)alloc((size_t)gB * gN * 64 * 2);   // (B, 64, N)
    bf16*  cwbT    = (bf16*)alloc((size_t)gB * gN * 32 * 2);   // (B, 32, N)
    bf16*  hbf     = (bf16*)alloc((size_t)gB * gN * 64 * 2);   // (B, N, 64) row-major
    float* h1f     = (float*)alloc((size_t)gB * gN * 64 * 4);
    float* hf      = (float*)alloc((size_t)gB * gN * 64 * 4);
    float* c1f     = (float*)alloc((size_t)gB * gN * 32 * 4);
    float* c2f     = (float*)alloc((size_t)gB * gN * gNC * 4);
    int*   ids     = (int*)alloc((size_t)gB * gN * 4);
    float* mu      = (float*)alloc((size_t)gB * 64 * 4);
    float* istd    = (float*)alloc((size_t)gB * 64 * 4);
    float* sums    = (float*)alloc((size_t)gB * gNC * 4);
    float* cnts    = (float*)alloc((size_t)gB * gNC * 4);
    int*   topbuf  = (int*)alloc((size_t)gB * 2 * 4);
    float* outvec  = (float*)alloc((size_t)gB * gN * 4);
    float* mlp1out = (float*)alloc((size_t)gB * gDMID * 4);
    (void)ws_size; (void)n_in; (void)in_sizes; (void)out_size;

    zero_kernel<<<1, 256, 0, stream>>>(sums, cnts, gB * gNC);

    // adj = relu(a1+a2+a3)
    long tot4 = (long)gB * gN * gN / 4;
    adj_relu_kernel<<<(unsigned)((tot4 + 255) / 256), 256, 0, stream>>>(a1, a2, a3, adj, tot4);

    // XW1 = x @ gc1_w  (bf16, transposed)
    long t64 = (long)gB * gN * 64;
    small_gemm_kernel<<<(unsigned)((t64 + 255) / 256), 256, 0, stream>>>(
        x, gc1_w, nullptr, nullptr, xw1bT, gN, gNFEAT, 64, 0);

    // h1 = relu(adj @ XW1 + gc1_b)   [WMMA]
    dim3 g64(gN / 64, gB);
    adj_gemm_kernel<64, 64><<<g64, 256, 0, stream>>>(adj, xw1bT, gc1_b, h1f, 1);

    // HW2 = h1 @ gc2_w (bf16, transposed)
    small_gemm_kernel<<<(unsigned)((t64 + 255) / 256), 256, 0, stream>>>(
        h1f, gc2_w, nullptr, nullptr, hw2bT, gN, 64, 64, 0);

    // h = adj @ HW2 + gc2_b          [WMMA]
    adj_gemm_kernel<64, 64><<<g64, 256, 0, stream>>>(adj, hw2bT, gc2_b, hf, 0);

    // log_softmax over class dim; BN over nodes -> h (f32 + bf16 row-major)
    long tn = (long)gB * gN;
    log_softmax_kernel<<<(unsigned)((tn + 255) / 256), 256, 0, stream>>>(hf, gN, 64);
    dim3 gs64(64, gB);
    bn_stats_kernel<<<gs64, 256, 0, stream>>>(hf, mu, istd, gN, 64);
    bn_apply_kernel<<<(unsigned)((t64 + 255) / 256), 256, 0, stream>>>(
        hf, hbf, mu, istd, norm_g, norm_b, gN, 64);

    // CW = h @ clgc_w (bf16, transposed)
    long t32 = (long)gB * gN * 32;
    small_gemm_kernel<<<(unsigned)((t32 + 255) / 256), 256, 0, stream>>>(
        hf, clgc_w, nullptr, nullptr, cwbT, gN, 64, 32, 0);

    // c1 = relu(adj @ CW + clgc_b)   [WMMA]
    dim3 g128(gN / 128, gB);
    adj_gemm_kernel<128, 32><<<g128, 256, 0, stream>>>(adj, cwbT, clgc_b, c1f, 1);

    // c2 = relu(c1 @ clmlp_w + clmlp_b); BN over nodes
    long t25 = (long)gB * gN * gNC;
    small_gemm_kernel<<<(unsigned)((t25 + 255) / 256), 256, 0, stream>>>(
        c1f, clmlp_w, clmlp_b, c2f, nullptr, gN, 32, gNC, 1);
    dim3 gs25(gNC, gB);
    bn_stats_kernel<<<gs25, 256, 0, stream>>>(c2f, mu, istd, gN, gNC);
    bn_apply_kernel<<<(unsigned)((t25 + 255) / 256), 256, 0, stream>>>(
        c2f, nullptr, mu, istd, cln_g, cln_b, gN, gNC);

    // argmax / centrality / segment sums; scores + top-2; mask
    cluster_kernel<<<(unsigned)((tn + 255) / 256), 256, 0, stream>>>(c2f, cen_w, cen_b, ids, sums, cnts);
    score_top_kernel<<<gB, 32, 0, stream>>>(sums, cnts, Cs, cid, topbuf);
    mask_kernel<<<(unsigned)((tn + 255) / 256), 256, 0, stream>>>(ids, topbuf, Tm);

    // out[b,n] = sum_m relu(<h_n,h_m>) * state[b,m]   [WMMA, fused]
    hht_state_kernel<<<g128, 256, 0, stream>>>(hbf, state, outvec);

    // BN over batch; MLP1; BN; MLP2 + sigmoid
    bn_batch_kernel<<<(gN + 255) / 256, 256, 0, stream>>>(outvec, norm1_g, norm1_b, gN);
    long tm1 = (long)gB * gDMID;
    mlp_kernel<<<(unsigned)((tm1 + 255) / 256), 256, 0, stream>>>(
        outvec, mlp1_w, mlp1_b, mlp1out, gN, gDMID, 1);
    bn_batch_kernel<<<(gDMID + 255) / 256, 256, 0, stream>>>(mlp1out, norm2_g, norm2_b, gDMID);
    long tm2 = (long)gB * 2 * gN;
    mlp_kernel<<<(unsigned)((tm2 + 255) / 256), 256, 0, stream>>>(
        mlp1out, mlp2_w, mlp2_b, outp, gDMID, 2 * gN, 2);
}
